// GCNEncoder_86535001080053
// MI455X (gfx1250) — compile-verified, matrix operand order checked by reading, not run on hardware
//
#include <hip/hip_runtime.h>

// ---------------------------------------------------------------------------
// GCN + TransformerEncoderLayer forward for MI455X (gfx1250, wave32, WMMA).
// All GEMMs use v_wmma_f32_16x16x32_bf16 (bf16 operands, fp32 accumulate).
// GEMM k-loop is unrolled x2 with ping-pong register sets: no rotation copies,
// no WMMA->VALU hazard nops, loads overlap wmma via partial s_wait_loadcnt.
// Attention is flash-style with V-fragment loads hoisted over softmax VALU.
// ---------------------------------------------------------------------------

#define N_NODES 4096
#define E_EDGES 131072
#define DIN     512
#define DMODEL  256
#define DFF     2048
#define DHEAD   128
#define QKV_DIM 768   // 3 * DMODEL

typedef __bf16 bf16_t;
typedef __attribute__((ext_vector_type(16))) __bf16 v16bf;
typedef __attribute__((ext_vector_type(8)))  __bf16 v8bf;
typedef __attribute__((ext_vector_type(8)))  float  v8f;

// float -> bf16, round-to-nearest-even (bit exact, no reliance on native cvt)
__device__ __forceinline__ bf16_t f2bf(float f) {
  union { float f; unsigned u; } v; v.f = f;
  unsigned r = v.u + 0x7FFFu + ((v.u >> 16) & 1u);
  unsigned short h = (unsigned short)(r >> 16);
  return __builtin_bit_cast(bf16_t, h);
}

// Load one 16x32 bf16 WMMA A/B fragment from a row-major matrix where the
// fragment's 32-wide dimension is contiguous (stride 1) in memory.
// Per ISA layout: lane holds row (lane&15); half index i maps to
//   k = (i&7) + 8*((i>>3)*2 | (lane>>4))
// so each lane's 16 values are two contiguous runs of 8 -> two b128 loads.
__device__ __forceinline__ v16bf load_frag(const bf16_t* base, int row, int ld,
                                           int k0, int lane) {
  int hi = (lane >> 4) & 1;
  const bf16_t* p = base + (long)row * ld + k0 + 8 * hi;
  v8bf lo = *(const v8bf*)(p);
  v8bf up = *(const v8bf*)(p + 16);
  v16bf f;
#pragma unroll
  for (int i = 0; i < 8; ++i) { f[i] = lo[i]; f[i + 8] = up[i]; }
  return f;
}

#define WMMA_BF16(a, b, c) \
  __builtin_amdgcn_wmma_f32_16x16x32_bf16(false, (a), false, (b), (short)0, (c), false, false)

// ---------------------------------------------------------------------------
// Small utility kernels
// ---------------------------------------------------------------------------
__global__ void fill_f32_kernel(float* p, float v, int n) {
  int i = blockIdx.x * blockDim.x + threadIdx.x;
  if (i < n) p[i] = v;
}

__global__ void cvt_bf16_kernel(const float* in, bf16_t* out, int n) {
  int i = blockIdx.x * blockDim.x + threadIdx.x;
  if (i < n) out[i] = f2bf(in[i]);
}

// out[c][r] = in[r][c]  (used to turn W_gcn [K,N] into [N,K] row-along-k form)
__global__ void transpose_bf16_kernel(const float* in, bf16_t* out,
                                      int rows, int cols) {
  int i = blockIdx.x * blockDim.x + threadIdx.x;
  if (i < rows * cols) {
    int r = i / cols, c = i % cols;
    out[(long)c * rows + r] = f2bf(in[i]);
  }
}

// deg[dst] += w for edges, +1 self loops
__global__ void deg_kernel(const int* ei, const float* ew, float* deg) {
  int i = blockIdx.x * blockDim.x + threadIdx.x;
  if (i < E_EDGES + N_NODES) {
    int dst; float w;
    if (i < E_EDGES) { dst = ei[E_EDGES + i]; w = ew[i]; }
    else             { dst = i - E_EDGES;     w = 1.0f; }
    atomicAdd(&deg[dst], w);
  }
}

__global__ void dinv_kernel(float* deg) {
  int i = blockIdx.x * blockDim.x + threadIdx.x;
  if (i < N_NODES) {
    float d = deg[i];
    deg[i] = d > 0.0f ? rsqrtf(d) : 0.0f;
  }
}

// one edge per 256-thread block; agg[dst, :] += norm * xw[src, :]
__global__ void scatter_kernel(const int* ei, const float* ew, const float* dinv,
                               const float* xw, float* agg) {
  int e = blockIdx.x, t = threadIdx.x;
  int s, d; float w;
  if (e < E_EDGES) { s = ei[e]; d = ei[E_EDGES + e]; w = ew[e]; }
  else             { s = d = e - E_EDGES;            w = 1.0f;  }
  float norm = dinv[s] * w * dinv[d];
  atomicAdd(&agg[(long)d * DMODEL + t], norm * xw[(long)s * DMODEL + t]);
}

// h = relu(agg + b_gcn), write fp32 + bf16
__global__ void bias_relu_kernel(const float* agg, const float* b,
                                 float* h, bf16_t* hbf) {
  int i = blockIdx.x * blockDim.x + threadIdx.x;
  if (i < N_NODES * DMODEL) {
    float v = fmaxf(agg[i] + b[i % DMODEL], 0.0f);
    h[i] = v;
    hbf[i] = f2bf(v);
  }
}

// split qkv fp32 [N,768] -> q (bf16, pre-scaled by 1/sqrt(dh)), k (bf16),
// v transposed to [256][N] bf16 so the P@V B-fragment is contiguous.
__global__ void qkv_split_kernel(const float* qkv, bf16_t* qbf, bf16_t* kbf,
                                 bf16_t* vt) {
  const float SCALE = 0.08838834764831845f; // 1/sqrt(128)
  int i = blockIdx.x * blockDim.x + threadIdx.x;
  if (i < N_NODES * QKV_DIM) {
    int n = i / QKV_DIM, c = i % QKV_DIM;
    float v = qkv[i];
    if (c < DMODEL)            qbf[(long)n * DMODEL + c] = f2bf(v * SCALE);
    else if (c < 2 * DMODEL)   kbf[(long)n * DMODEL + (c - DMODEL)] = f2bf(v);
    else                       vt[(long)(c - 2 * DMODEL) * N_NODES + n] = f2bf(v);
  }
}

// out = LayerNorm(a + b) * g + beta ; optional bf16 copy
__global__ void ln_kernel(const float* a, const float* b, const float* g,
                          const float* beta, float* outf, bf16_t* outbf) {
  __shared__ float s1[DMODEL], s2[DMODEL];
  int row = blockIdx.x, t = threadIdx.x;
  float v = a[(long)row * DMODEL + t] + b[(long)row * DMODEL + t];
  s1[t] = v; s2[t] = v * v;
  __syncthreads();
  for (int st = DMODEL / 2; st > 0; st >>= 1) {
    if (t < st) { s1[t] += s1[t + st]; s2[t] += s2[t + st]; }
    __syncthreads();
  }
  float mu  = s1[0] * (1.0f / DMODEL);
  float var = s2[0] * (1.0f / DMODEL) - mu * mu;
  float r   = rsqrtf(var + 1e-5f);
  float o   = (v - mu) * r * g[t] + beta[t];
  outf[(long)row * DMODEL + t] = o;
  if (outbf) outbf[(long)row * DMODEL + t] = f2bf(o);
}

// ---------------------------------------------------------------------------
// Generic bf16 WMMA GEMM:  C[M,N] = A[M,K] @ B[N,K]^T (+bias, +relu)
// Block = 4 waves; each wave computes a 32x64 tile (2 A frags x 4 B frags ->
// 8 wmma per k-step, 6 fragment loads). k-loop unrolled x2 with ping-pong
// register sets (no rotation copies). Requires K % 64 == 0.
// ---------------------------------------------------------------------------
__device__ __forceinline__ void gemm_load_step(
    const bf16_t* __restrict__ A, const bf16_t* __restrict__ B,
    int rowA0, int rowA1, int nb, int K, int k0, int lane,
    v16bf (&a)[2], v16bf (&b)[4]) {
  a[0] = load_frag(A, rowA0, K, k0, lane);
  a[1] = load_frag(A, rowA1, K, k0, lane);
#pragma unroll
  for (int t = 0; t < 4; ++t)
    b[t] = load_frag(B, nb + t * 16, K, k0, lane);
}

__device__ __forceinline__ void gemm_mma_step(
    const v16bf (&a)[2], const v16bf (&b)[4], v8f (&acc)[2][4]) {
#pragma unroll
  for (int t = 0; t < 4; ++t) {
    acc[0][t] = WMMA_BF16(a[0], b[t], acc[0][t]);
    acc[1][t] = WMMA_BF16(a[1], b[t], acc[1][t]);
  }
}

__global__ __launch_bounds__(128) void gemm_bf16_kernel(
    const bf16_t* __restrict__ A, const bf16_t* __restrict__ B,
    const float* __restrict__ bias, float* __restrict__ Cf,
    bf16_t* __restrict__ Cbf, int M, int N, int K, int relu) {
  int wave = threadIdx.x >> 5;
  int lane = threadIdx.x & 31;
  int m0 = blockIdx.y * 128 + wave * 32;
  int n0 = blockIdx.x * 64;
  int rl = lane & 15;
  int rowA0 = m0 + rl;
  int rowA1 = m0 + 16 + rl;
  int nb = n0 + rl;

  v8f acc[2][4] = {};
  v16bf a0[2], b0[4], a1[2], b1[4];

  // steps = K/32 (even at every call site); set0 holds step s invariantly
  gemm_load_step(A, B, rowA0, rowA1, nb, K, 0, lane, a0, b0);
  int k0 = 0;
  for (; k0 + 64 < K; k0 += 64) {
    gemm_load_step(A, B, rowA0, rowA1, nb, K, k0 + 32, lane, a1, b1);
    gemm_mma_step(a0, b0, acc);
    gemm_load_step(A, B, rowA0, rowA1, nb, K, k0 + 64, lane, a0, b0);
    gemm_mma_step(a1, b1, acc);
  }
  // epilogue: set0 holds step K/32-2; load last step, two mma groups
  gemm_load_step(A, B, rowA0, rowA1, nb, K, k0 + 32, lane, a1, b1);
  gemm_mma_step(a0, b0, acc);
  gemm_mma_step(a1, b1, acc);

  int half = (lane >> 4) & 1;
#pragma unroll
  for (int rt = 0; rt < 2; ++rt) {
#pragma unroll
    for (int t = 0; t < 4; ++t) {
#pragma unroll
      for (int v = 0; v < 8; ++v) {
        int r = m0 + rt * 16 + v + 8 * half;
        int c = n0 + t * 16 + rl;
        float val = acc[rt][t][v];
        if (bias) val += bias[c];
        if (relu) val = fmaxf(val, 0.0f);
        long idx = (long)r * N + c;
        if (Cf)  Cf[idx]  = val;
        if (Cbf) Cbf[idx] = f2bf(val);
      }
    }
  }
}

// ---------------------------------------------------------------------------
// Flash attention, 1 wave per (16-query block, head). dh = 128, keys in
// blocks of 32 so P (16x32) is exactly one bf16 A-fragment (repacked via LDS).
// V fragments are loaded before the softmax VALU block so their global
// latency overlaps the shuffle/exp work.
// ---------------------------------------------------------------------------
__global__ __launch_bounds__(32) void attn_kernel(
    const bf16_t* __restrict__ qbf, const bf16_t* __restrict__ kbf,
    const bf16_t* __restrict__ vt, float* __restrict__ o,
    bf16_t* __restrict__ obf) {
  __shared__ __align__(16) bf16_t pS[16 * 32];
  int lane = threadIdx.x;
  int qblk = blockIdx.x;
  int head = blockIdx.y;
  int hoff = head * DHEAD;
  int cl = lane & 15;
  int half = (lane >> 4) & 1;

  v16bf qf[4];
#pragma unroll
  for (int c = 0; c < 4; ++c)
    qf[c] = load_frag(qbf, qblk * 16 + cl, DMODEL, hoff + c * 32, lane);

  v8f oacc[8] = {};
  float m[8], l[8];
#pragma unroll
  for (int v = 0; v < 8; ++v) { m[v] = -1e30f; l[v] = 0.0f; }

  for (int j = 0; j < N_NODES; j += 32) {
    // S = Q (16x128) @ K^T (128x32): two 16x16 C tiles, 4 k-chunks each
    v8f s0 = {}, s1 = {};
#pragma unroll
    for (int c = 0; c < 4; ++c) {
      v16bf k0 = load_frag(kbf, j + cl,      DMODEL, hoff + c * 32, lane);
      v16bf k1 = load_frag(kbf, j + 16 + cl, DMODEL, hoff + c * 32, lane);
      s0 = WMMA_BF16(qf[c], k0, s0);
      s1 = WMMA_BF16(qf[c], k1, s1);
    }
    // V fragments for this key block: independent of softmax -> load early
    v16bf vf[8];
#pragma unroll
    for (int dc = 0; dc < 8; ++dc)
      vf[dc] = load_frag(vt, hoff + dc * 16 + cl, N_NODES, j, lane);

    // online softmax (row stats: row = v + 8*half, reduce over 16-lane half)
    float alpha[8];
#pragma unroll
    for (int v = 0; v < 8; ++v) {
      float rm = fmaxf(s0[v], s1[v]);
#pragma unroll
      for (int d = 1; d < 16; d <<= 1) rm = fmaxf(rm, __shfl_xor(rm, d, 32));
      float mn = fmaxf(m[v], rm);
      float a  = __expf(m[v] - mn);
      float p0 = __expf(s0[v] - mn);
      float p1 = __expf(s1[v] - mn);
      float rs = p0 + p1;
#pragma unroll
      for (int d = 1; d < 16; d <<= 1) rs += __shfl_xor(rs, d, 32);
      l[v] = l[v] * a + rs;
      m[v] = mn;
      alpha[v] = a;
      int r = v + 8 * half;            // C-layout -> (row, col) in LDS tile
      pS[r * 32 + cl]      = f2bf(p0);
      pS[r * 32 + 16 + cl] = f2bf(p1);
    }
    __syncthreads();
    v16bf pf = load_frag(pS, cl, 32, 0, lane);   // P as A-fragment
#pragma unroll
    for (int dc = 0; dc < 8; ++dc) {
#pragma unroll
      for (int v = 0; v < 8; ++v) oacc[dc][v] *= alpha[v];
      oacc[dc] = WMMA_BF16(pf, vf[dc], oacc[dc]);
    }
    __syncthreads();
  }
#pragma unroll
  for (int dc = 0; dc < 8; ++dc) {
#pragma unroll
    for (int v = 0; v < 8; ++v) {
      int r = qblk * 16 + v + 8 * half;
      int c = hoff + dc * 16 + cl;
      float val = oacc[dc][v] / l[v];
      o[(long)r * DMODEL + c]   = val;
      obf[(long)r * DMODEL + c] = f2bf(val);
    }
  }
}

// ---------------------------------------------------------------------------
// Host-side orchestration
// ---------------------------------------------------------------------------
extern "C" void kernel_launch(void* const* d_in, const int* in_sizes, int n_in,
                              void* d_out, int out_size, void* d_ws,
                              size_t ws_size, hipStream_t stream) {
  const float* x       = (const float*)d_in[0];
  const int*   ei      = (const int*)  d_in[1];
  const float* ew      = (const float*)d_in[2];
  const float* Wgcn    = (const float*)d_in[3];
  const float* bgcn    = (const float*)d_in[4];
  const float* inw     = (const float*)d_in[5];
  const float* inb     = (const float*)d_in[6];
  const float* outw    = (const float*)d_in[7];
  const float* outb    = (const float*)d_in[8];
  const float* l1w     = (const float*)d_in[9];
  const float* l1b     = (const float*)d_in[10];
  const float* l2w     = (const float*)d_in[11];
  const float* l2b     = (const float*)d_in[12];
  const float* ln1g    = (const float*)d_in[13];
  const float* ln1b    = (const float*)d_in[14];
  const float* ln2g    = (const float*)d_in[15];
  const float* ln2b    = (const float*)d_in[16];
  float* out = (float*)d_out;

  // workspace layout (256B aligned slots), ~62 MB total
  char* w = (char*)d_ws;
  size_t off = 0;
  auto alloc = [&](size_t bytes) -> char* {
    char* p = w + off;
    off += (bytes + 255) & ~(size_t)255;
    return p;
  };
  bf16_t* xbf    = (bf16_t*)alloc((size_t)N_NODES * DIN * 2);
  bf16_t* wgcnT  = (bf16_t*)alloc((size_t)DMODEL * DIN * 2);
  bf16_t* inwbf  = (bf16_t*)alloc((size_t)QKV_DIM * DMODEL * 2);
  bf16_t* outwbf = (bf16_t*)alloc((size_t)DMODEL * DMODEL * 2);
  bf16_t* l1wbf  = (bf16_t*)alloc((size_t)DFF * DMODEL * 2);
  bf16_t* l2wbf  = (bf16_t*)alloc((size_t)DMODEL * DFF * 2);
  float*  deg    = (float*) alloc((size_t)N_NODES * 4);
  float*  xw     = (float*) alloc((size_t)N_NODES * DMODEL * 4);
  float*  agg    = (float*) alloc((size_t)N_NODES * DMODEL * 4);
  float*  h      = (float*) alloc((size_t)N_NODES * DMODEL * 4);
  bf16_t* hbf    = (bf16_t*)alloc((size_t)N_NODES * DMODEL * 2);
  float*  qkv    = (float*) alloc((size_t)N_NODES * QKV_DIM * 4);   // \ reused
  bf16_t* qbf    = (bf16_t*)alloc((size_t)N_NODES * DMODEL * 2);    //  | as
  bf16_t* kbf    = (bf16_t*)alloc((size_t)N_NODES * DMODEL * 2);    // /  ff1bf
  bf16_t* vt     = (bf16_t*)alloc((size_t)DMODEL * N_NODES * 2);
  float*  o      = (float*) alloc((size_t)N_NODES * DMODEL * 4);
  bf16_t* obf    = (bf16_t*)alloc((size_t)N_NODES * DMODEL * 2);
  float*  aproj  = (float*) alloc((size_t)N_NODES * DMODEL * 4);
  float*  h1     = (float*) alloc((size_t)N_NODES * DMODEL * 4);
  bf16_t* h1bf   = (bf16_t*)alloc((size_t)N_NODES * DMODEL * 2);
  float*  ff2    = (float*) alloc((size_t)N_NODES * DMODEL * 4);
  bf16_t* ff1bf  = (bf16_t*)qkv;  // 16 MB overlay on dead qkv/qbf/kbf

  const int TB = 256;
  auto blocks = [](int n, int t) { return (n + t - 1) / t; };

  // --- precision conversions ---
  cvt_bf16_kernel<<<blocks(N_NODES * DIN, TB), TB, 0, stream>>>(x, xbf, N_NODES * DIN);
  transpose_bf16_kernel<<<blocks(DIN * DMODEL, TB), TB, 0, stream>>>(Wgcn, wgcnT, DIN, DMODEL);
  cvt_bf16_kernel<<<blocks(QKV_DIM * DMODEL, TB), TB, 0, stream>>>(inw, inwbf, QKV_DIM * DMODEL);
  cvt_bf16_kernel<<<blocks(DMODEL * DMODEL, TB), TB, 0, stream>>>(outw, outwbf, DMODEL * DMODEL);
  cvt_bf16_kernel<<<blocks(DFF * DMODEL, TB), TB, 0, stream>>>(l1w, l1wbf, DFF * DMODEL);
  cvt_bf16_kernel<<<blocks(DMODEL * DFF, TB), TB, 0, stream>>>(l2w, l2wbf, DMODEL * DFF);

  // --- GCN: degrees, norm, xw = x @ Wgcn, scatter, bias+relu ---
  fill_f32_kernel<<<blocks(N_NODES, TB), TB, 0, stream>>>(deg, 0.0f, N_NODES);
  fill_f32_kernel<<<blocks(N_NODES * DMODEL, TB), TB, 0, stream>>>(agg, 0.0f, N_NODES * DMODEL);
  deg_kernel<<<blocks(E_EDGES + N_NODES, TB), TB, 0, stream>>>(ei, ew, deg);
  dinv_kernel<<<blocks(N_NODES, TB), TB, 0, stream>>>(deg);
  gemm_bf16_kernel<<<dim3(DMODEL / 64, N_NODES / 128), 128, 0, stream>>>(
      xbf, wgcnT, nullptr, xw, nullptr, N_NODES, DMODEL, DIN, 0);
  scatter_kernel<<<E_EDGES + N_NODES, DMODEL, 0, stream>>>(ei, ew, deg, xw, agg);
  bias_relu_kernel<<<blocks(N_NODES * DMODEL, TB), TB, 0, stream>>>(agg, bgcn, h, hbf);

  // --- MHA ---
  gemm_bf16_kernel<<<dim3(QKV_DIM / 64, N_NODES / 128), 128, 0, stream>>>(
      hbf, inwbf, inb, qkv, nullptr, N_NODES, QKV_DIM, DMODEL, 0);
  qkv_split_kernel<<<blocks(N_NODES * QKV_DIM, TB), TB, 0, stream>>>(qkv, qbf, kbf, vt);
  attn_kernel<<<dim3(N_NODES / 16, 2), 32, 0, stream>>>(qbf, kbf, vt, o, obf);
  gemm_bf16_kernel<<<dim3(DMODEL / 64, N_NODES / 128), 128, 0, stream>>>(
      obf, outwbf, outb, aproj, nullptr, N_NODES, DMODEL, DMODEL, 0);
  ln_kernel<<<N_NODES, DMODEL, 0, stream>>>(h, aproj, ln1g, ln1b, h1, h1bf);

  // --- FFN ---
  gemm_bf16_kernel<<<dim3(DFF / 64, N_NODES / 128), 128, 0, stream>>>(
      h1bf, l1wbf, l1b, nullptr, ff1bf, N_NODES, DFF, DMODEL, 1);
  gemm_bf16_kernel<<<dim3(DMODEL / 64, N_NODES / 128), 128, 0, stream>>>(
      ff1bf, l2wbf, l2b, ff2, nullptr, N_NODES, DMODEL, DFF, 0);
  ln_kernel<<<N_NODES, DMODEL, 0, stream>>>(h1, ff2, ln2g, ln2b, out, nullptr);
}
